// SAKT_24970939859703
// MI455X (gfx1250) — matrix operand order repaired and checked by reading, not compiled, not run on hardware
//
#include <hip/hip_runtime.h>
#include <hip/hip_bf16.h>

typedef __bf16 bf16_t;
typedef __attribute__((ext_vector_type(16))) __bf16 v16bf;
typedef __attribute__((ext_vector_type(8)))  float  v8f;
typedef unsigned int u32x4 __attribute__((ext_vector_type(4)));
typedef int i32x8 __attribute__((ext_vector_type(8)));
typedef int i32x4 __attribute__((ext_vector_type(4)));

#define WMMA_BF16(a,b,c) __builtin_amdgcn_wmma_f32_16x16x32_bf16(false,(a),false,(b),(short)0,(c),false,false)

constexpr int B_   = 16;
constexpr int L_   = 1024;
constexpr int S_   = 1023;
constexpr int SP   = 1024;   // padded sequence
constexpr int D_   = 768;
constexpr int H_   = 8;
constexpr int DH_  = 512;
constexpr int HID_ = 256;
constexpr int FF_  = 3072;
constexpr int NA_  = 10000;
constexpr int EHD  = D_ + HID_;  // 1024
constexpr int HDH  = H_ * DH_;   // 4096

union ABu { v16bf v; u32x4 q[2]; };

// A-fragment (16x32 bf16, row-major, stride lda). Layout (matches B fragment):
// lane half=0 holds K = 0..15, half=1 holds K = 16..31, row = lane&15.
__device__ __forceinline__ v16bf load_afrag(const bf16_t* base, int lda, int k0, int lane) {
  int r = lane & 15, half = lane >> 4;
  const bf16_t* p = base + (long)r * lda + k0 + half * 16;
  ABu u;
  u.q[0] = *(const u32x4*)(p);
  u.q[1] = *(const u32x4*)(p + 8);
  return u.v;
}

// B-fragment (32x16 bf16) loaded from BT (N x K row-major, stride ldb).
// lane n (0..15) holds column n with K=0..15; lane n+16 holds K=16..31.
__device__ __forceinline__ v16bf load_bfrag(const bf16_t* bt, int ldb, int k0, int lane) {
  int n = lane & 15;
  int kb = k0 + (lane >> 4) * 16;
  const bf16_t* p = bt + (long)n * ldb + kb;
  ABu u;
  u.q[0] = *(const u32x4*)(p);
  u.q[1] = *(const u32x4*)(p + 8);
  return u.v;
}

// ---------------- TDM: issue a 2D bf16 tile load (tile_h rows x tile_w elems) ----------------
// Descriptor packing per CDNA5 ISA sect. 8.3/8.4.
// Group0: count=1, lds_addr[63:32], global_addr[120:64], type=2.
// Group1: data_size=1 (2B), tensor_dim0=tile_w, tensor_dim1=tile_h (exact fit),
//         tile_dim0=tile_w, tile_dim1=tile_h, tensor_dim0_stride=row_stride.
__device__ __forceinline__ void tdm_load_2d(unsigned lds_byte, const bf16_t* gaddr,
                                            unsigned tile_w, unsigned tile_h,
                                            long row_stride_elems) {
  unsigned long long ga = (unsigned long long)(__UINTPTR_TYPE__)gaddr;
  u32x4 g0 = {};
  g0.x = 1u;                                                  // count=1 valid user descriptor
  g0.y = lds_byte;                                            // lds_addr (bytes)
  g0.z = (unsigned)(ga & 0xffffffffu);                        // global_addr[31:0]
  g0.w = (unsigned)((ga >> 32) & 0x1ffffffu) | (2u << 30);    // global_addr[56:32] | type=2
  unsigned long long str = (unsigned long long)row_stride_elems;
  i32x8 g1 = {};
  g1[0] = 0x10000;                                            // data_size=1 (2 bytes)
  g1[1] = (int)((tile_w & 0xffffu) << 16);                    // tensor_dim0[15:0] @ bits 63:48
  g1[2] = (int)((tile_w >> 16) | ((tile_h & 0xffffu) << 16)); // dim0 hi | tensor_dim1 lo
  g1[3] = (int)((tile_h >> 16) | ((tile_w & 0xffffu) << 16)); // dim1 hi | tile_dim0
  g1[4] = (int)(tile_h & 0xffffu);                            // tile_dim1 (tile_dim2 = 0)
  g1[5] = (int)(str & 0xffffffffu);                           // tensor_dim0_stride[31:0]
  g1[6] = (int)((str >> 32) & 0xffffu);                       // tensor_dim0_stride[47:32]
  g1[7] = 0;
  i32x4 zz = {};
#if __has_include(<hip/amd_detail/amd_gfx1250_TDM.h>)
  i32x8 zz8 = {};
  __builtin_amdgcn_tensor_load_to_lds(g0, g1, zz, zz, zz8, 0);   // clang-23 / therock 6-arg
#else
  __builtin_amdgcn_tensor_load_to_lds(g0, g1, zz, zz, 0);        // ROCm 7.2 5-arg
#endif
}

// ---------------- prep: f32 -> bf16 transpose (out[n][k] = in[k][n]) ----------------
__global__ void k_transpose_bf16(const float* __restrict__ in, bf16_t* __restrict__ out,
                                 int Kd, int Nd) {
  long batch = blockIdx.z;
  long total = (long)Kd * Nd;
  in  += batch * total;
  out += batch * total;
  long i = (long)blockIdx.x * 256 + threadIdx.x;
  if (i >= total) return;
  long n = i / Kd, k = i % Kd;
  out[i] = (bf16_t)in[k * (long)Nd + n];
}

// ---------------- build M_hat (B,SP,D) bf16 ----------------
__global__ void k_build_mhat(const int* __restrict__ cate, const int* __restrict__ targ,
                             const float* __restrict__ Memb, const float* __restrict__ pos,
                             bf16_t* __restrict__ Mhat) {
  long i = (long)blockIdx.x * 256 + threadIdx.x;
  const long total = (long)B_ * SP * D_;
  if (i >= total) return;
  int d  = (int)(i % D_);
  long bs = i / D_;
  int s = (int)(bs % SP);
  int b = (int)(bs / SP);
  float v = 0.f;
  if (s < S_) {
    int y = cate[((long)b * L_ + s) * 4] + targ[(long)b * L_ + s] * NA_;
    v = Memb[(long)y * D_ + d] + pos[(long)s * D_ + d];
  }
  Mhat[i] = (bf16_t)v;
}

// ---------------- build E_hat (B,SP,EHD) bf16 ----------------
__global__ void k_build_ehat(const int* __restrict__ cate, const float* __restrict__ contx,
                             const float* __restrict__ Eemb, const float* __restrict__ cemb,
                             const float* __restrict__ cW, const float* __restrict__ cb,
                             bf16_t* __restrict__ Ehat) {
  long i = (long)blockIdx.x * 256 + threadIdx.x;
  const long total = (long)B_ * SP * EHD;
  if (i >= total) return;
  int d  = (int)(i % EHD);
  long bs = i / EHD;
  int s = (int)(bs % SP);
  int b = (int)(bs / SP);
  float v = 0.f;
  if (s < S_) {
    int cx = cate[((long)b * L_ + (s + 1)) * 4];
    if (d < D_) {
      v = Eemb[(long)cx * D_ + d];
    } else {
      int j = d - D_;
      v = cemb[(long)cx * HID_ + j] + cb[j];
      const float* cp = contx + ((long)b * L_ + (s + 1)) * 6;
#pragma unroll
      for (int c = 0; c < 6; ++c) v += cp[c] * cW[c * HID_ + j];
    }
  }
  Ehat[i] = (bf16_t)v;
}

// ---------------- generic bf16 WMMA GEMM with TDM double-buffered LDS staging ----------------
// C[m][n] = sum_k A[m][k] * BT[n][k] (+bias[n]) (+resid) ; optional relu; dual f32/bf16 store.
// Block: 256 threads (8 waves), block tile 128(M) x 64(N), K stepped by 32.
// Wave 0 drives the Tensor Data Mover: A tile (128x32) + B tile (64x32) per step, 2 LDS buffers.
// vt != 0 : transposed bf16 store into V^T layout [z][b][n][s] with m = b*SP + s.
__global__ __launch_bounds__(256) void k_gemm(
    const bf16_t* __restrict__ A, const bf16_t* __restrict__ BT,
    const float* __restrict__ bias, const float* __restrict__ resid,
    float* __restrict__ Cf, bf16_t* __restrict__ Cb,
    int M, int N, int Kd, long sA, long sB, long sBias, long sC, int relu, int vt) {
  __shared__ __align__(16) bf16_t shA[2][128 * 32];
  __shared__ __align__(16) bf16_t shB[2][64 * 32];

  int lane = threadIdx.x & 31;
  int w    = threadIdx.x >> 5;           // 8 waves
  long z   = blockIdx.z;
  A  += z * sA;
  BT += z * sB;
  const float* biasp = bias ? bias + z * sBias : nullptr;
  int m0 = blockIdx.x * 128 + w * 16;    // all M here are multiples of 128
  int n0 = blockIdx.y * 64;

  const bf16_t* Ablk = A + (long)(blockIdx.x * 128) * Kd;
  const bf16_t* Bblk = BT + (long)n0 * Kd;
  const int steps = Kd / 32;
  const bool issuer = (w == 0);

  unsigned offA[2] = { (unsigned)(__UINTPTR_TYPE__)&shA[0][0],
                       (unsigned)(__UINTPTR_TYPE__)&shA[1][0] };
  unsigned offB[2] = { (unsigned)(__UINTPTR_TYPE__)&shB[0][0],
                       (unsigned)(__UINTPTR_TYPE__)&shB[1][0] };

  if (issuer) {
    tdm_load_2d(offA[0], Ablk, 32, 128, Kd);
    tdm_load_2d(offB[0], Bblk, 32, 64, Kd);
    if (steps > 1) {
      tdm_load_2d(offA[1], Ablk + 32, 32, 128, Kd);
      tdm_load_2d(offB[1], Bblk + 32, 32, 64, Kd);
    }
  }

  v8f acc[4] = {};
  for (int i = 0; i < steps; ++i) {
    if (issuer) {
      if (i + 1 < steps) __builtin_amdgcn_s_wait_tensorcnt(2);  // buf i done, buf i+1 in flight
      else               __builtin_amdgcn_s_wait_tensorcnt(0);
    }
    __syncthreads();                     // TDM-filled buffer visible to all waves

    const bf16_t* a_s = &shA[i & 1][0] + (w * 16) * 32;
    v16bf a = load_afrag(a_s, 32, 0, lane);
#pragma unroll
    for (int f = 0; f < 4; ++f) {
      v16bf b = load_bfrag(&shB[i & 1][0] + (f * 16) * 32, 32, 0, lane);
      acc[f] = WMMA_BF16(a, b, acc[f]);
    }
    __syncthreads();                     // all reads of buf (i&1) complete before refill

    if (issuer && (i + 2) < steps) {
      int kn = (i + 2) * 32;
      tdm_load_2d(offA[i & 1], Ablk + kn, 32, 128, Kd);
      tdm_load_2d(offB[i & 1], Bblk + kn, 32, 64, Kd);
    }
  }

  int r = lane & 15, half = lane >> 4;
#pragma unroll
  for (int f = 0; f < 4; ++f) {
    int n = n0 + f * 16 + r;
    float bv = biasp ? biasp[n] : 0.f;
#pragma unroll
    for (int i = 0; i < 8; ++i) {
      int m = m0 + half * 8 + i;
      float v = acc[f][i] + bv;
      if (relu) v = fmaxf(v, 0.f);
      long cidx = (long)m * N + n + z * sC;
      if (resid) v += resid[cidx];
      if (Cf) Cf[cidx] = v;
      if (Cb) {
        if (vt) {
          int bb = m >> 10, s = m & (SP - 1);   // SP = 1024
          long o = z * ((long)B_ * DH_ * SP) + (long)bb * DH_ * SP + (long)n * SP + s;
          Cb[o] = (bf16_t)v;
        } else {
          Cb[cidx] = (bf16_t)v;
        }
      }
    }
  }
}

// ---------------- fused flash attention ----------------
// grid = (SP/16, B, H), block = 128 (4 waves). Each block: 16 q rows, full DH=512.
// Wave w owns DH slice [w*128, w*128+128). k stepped by 32.
__global__ __launch_bounds__(128) void k_attn(const bf16_t* __restrict__ Q,
                                              const bf16_t* __restrict__ Km,
                                              const bf16_t* __restrict__ Vt,
                                              bf16_t* __restrict__ Zc) {
  const int qt = blockIdx.x, b = blockIdx.y, h = blockIdx.z;
  const int tid = threadIdx.x;
  const int lane = tid & 31, w = tid >> 5;
  const int r = lane & 15, half = lane >> 4;
  const float scale = 0.04419417382415922f;   // 1/sqrt(512)

  __shared__ __align__(16) float  part[4][16][32];
  __shared__ __align__(16) float  sS[16][32];
  __shared__ __align__(16) bf16_t P[16][32];
  __shared__ float rowmax[16], rowden[16], rfac[16];

  const bf16_t* Qp = Q  + (((long)h * B_ + b) * SP + qt * 16) * DH_;
  const bf16_t* Kp = Km + (((long)h * B_ + b) * SP) * DH_;
  const bf16_t* Vp = Vt + (((long)h * B_ + b) * DH_) * SP;

  v8f accZ[8] = {};
  if (tid < 16) { rowmax[tid] = -1e30f; rowden[tid] = 0.f; }
  __syncthreads();

  for (int kk0 = 0; kk0 < SP; kk0 += 32) {
    __builtin_prefetch(Kp + (long)(kk0 + 32) * DH_ + w * 128, 0, 1);  // next K panel
    // --- partial scores over this wave's 128-wide DH slice ---
    v8f aS[2] = {};
#pragma unroll
    for (int j = 0; j < 4; ++j) {
      int kd = w * 128 + j * 32;
      v16bf aq = load_afrag(Qp, DH_, kd, lane);
#pragma unroll
      for (int nh = 0; nh < 2; ++nh) {
        v16bf bk = load_bfrag(Kp + (long)(kk0 + nh * 16) * DH_, DH_, kd, lane);
        aS[nh] = WMMA_BF16(aq, bk, aS[nh]);
      }
    }
#pragma unroll
    for (int nh = 0; nh < 2; ++nh)
#pragma unroll
      for (int i = 0; i < 8; ++i)
        part[w][half * 8 + i][nh * 16 + r] = aS[nh][i];
    __syncthreads();

    // --- reduce partials, scale + mask ---
    for (int idx = tid; idx < 512; idx += 128) {
      int m = idx >> 5, n = idx & 31;
      float s = part[0][m][n] + part[1][m][n] + part[2][m][n] + part[3][m][n];
      int kg = kk0 + n;
      // reference: additive mask = 0 iff key index >= b-1 (triu((B,S),k=-1)); pad keys masked
      float msk = (kg < S_ && kg >= b - 1) ? 0.f : -100000.f;
      sS[m][n] = s * scale + msk;
    }
    __syncthreads();

    // --- online softmax bookkeeping (one thread per q row) ---
    if (tid < 16) {
      int m = tid;
      float mx = rowmax[m];
      for (int n = 0; n < 32; ++n) mx = fmaxf(mx, sS[m][n]);
      float f = __expf(rowmax[m] - mx);
      rfac[m] = f;
      float sum = 0.f;
      for (int n = 0; n < 32; ++n) {
        float p = __expf(sS[m][n] - mx);
        P[m][n] = (bf16_t)p;
        sum += p;
      }
      rowden[m] = rowden[m] * f + sum;
      rowmax[m] = mx;
    }
    __syncthreads();

    // --- rescale accumulators, then Z += P x V ---
    float fi[8];
#pragma unroll
    for (int i = 0; i < 8; ++i) fi[i] = rfac[half * 8 + i];
#pragma unroll
    for (int f = 0; f < 8; ++f)
#pragma unroll
      for (int i = 0; i < 8; ++i) accZ[f][i] = accZ[f][i] * fi[i];

    v16bf ap = load_afrag((const bf16_t*)&P[0][0], 32, 0, lane);
#pragma unroll
    for (int f = 0; f < 8; ++f) {
      const bf16_t* vp = Vp + (long)(w * 128 + f * 16 + r) * SP + kk0 + half * 16;
      ABu u;
      u.q[0] = *(const u32x4*)(vp);
      u.q[1] = *(const u32x4*)(vp + 8);
      accZ[f] = WMMA_BF16(ap, u.v, accZ[f]);
    }
    __syncthreads();
  }

  // --- normalize and store Zc[b][s][h*DH + dh] ---
#pragma unroll
  for (int f = 0; f < 8; ++f) {
#pragma unroll
    for (int i = 0; i < 8; ++i) {
      int m = half * 8 + i;
      int s = qt * 16 + m;
      if (s < S_) {
        float v = accZ[f][i] / rowden[m];
        Zc[((long)b * SP + s) * HDH + h * DH_ + w * 128 + f * 16 + r] = (bf16_t)v;
      }
    }
  }
}

// ---------------- LayerNorm + final projection ----------------
__device__ __forceinline__ float block_sum(float v, float* red) {
  int t = threadIdx.x;
  red[t] = v;
  __syncthreads();
  for (int off = 128; off > 0; off >>= 1) {
    if (t < off) red[t] += red[t + off];
    __syncthreads();
  }
  float r = red[0];
  __syncthreads();
  return r;
}

__global__ __launch_bounds__(256) void k_lnfinal(const float* __restrict__ hbuf,
                                                 const float* __restrict__ g,
                                                 const float* __restrict__ bt,
                                                 const float* __restrict__ finW,
                                                 const float* __restrict__ finb,
                                                 float* __restrict__ out) {
  __shared__ float red[256];
  int row = blockIdx.x;            // b*S + s
  int b = row / S_, s = row % S_;
  const float* hp = hbuf + ((long)b * SP + s) * D_;

  float ls = 0.f;
  for (int d = threadIdx.x; d < D_; d += 256) ls += hp[d];
  float mu = block_sum(ls, red) / (float)D_;

  ls = 0.f;
  for (int d = threadIdx.x; d < D_; d += 256) { float t = hp[d] - mu; ls += t * t; }
  float var = block_sum(ls, red) / (float)D_;
  float inv = rsqrtf(var + 1e-5f);

  ls = 0.f;
  for (int d = threadIdx.x; d < D_; d += 256) {
    float zn = (hp[d] - mu) * inv * g[d] + bt[d];
    ls += zn * finW[d];
  }
  float tot = block_sum(ls, red);
  if (threadIdx.x == 0) out[row] = tot + finb[0];
}

// ---------------- host launcher ----------------
extern "C" void kernel_launch(void* const* d_in, const int* in_sizes, int n_in,
                              void* d_out, int out_size, void* d_ws, size_t ws_size,
                              hipStream_t stream) {
  const int*   cate_x  = (const int*)  d_in[0];
  const float* cont_x  = (const float*)d_in[1];
  // d_in[2] = mask (unused by reference math)
  const int*   targets = (const int*)  d_in[3];
  const float* M_emb   = (const float*)d_in[4];
  const float* E_emb   = (const float*)d_in[5];
  const float* pos_emb = (const float*)d_in[6];
  const float* cate_emb= (const float*)d_in[7];
  const float* cont_W  = (const float*)d_in[8];
  const float* cont_b  = (const float*)d_in[9];
  const float* WQ      = (const float*)d_in[10];
  const float* bQ      = (const float*)d_in[11];
  const float* WK      = (const float*)d_in[12];
  const float* bK      = (const float*)d_in[13];
  const float* WV      = (const float*)d_in[14];
  const float* bV      = (const float*)d_in[15];
  const float* W0      = (const float*)d_in[16];
  const float* ln_g    = (const float*)d_in[17];
  const float* ln_b    = (const float*)d_in[18];
  const float* ffn_W1  = (const float*)d_in[19];
  const float* ffn_b1  = (const float*)d_in[20];
  const float* ffn_W2  = (const float*)d_in[21];
  const float* ffn_b2  = (const float*)d_in[22];
  const float* fin_W   = (const float*)d_in[23];
  const float* fin_b   = (const float*)d_in[24];
  float* out = (float*)d_out;

  char* wp = (char*)d_ws;
  auto alloc = [&](size_t bytes) -> char* {
    char* p = wp;
    wp += (bytes + 255) & ~(size_t)255;
    return p;
  };

  const long MROWS = (long)B_ * SP;   // 16384
  bf16_t* Mhat = (bf16_t*)alloc((size_t)MROWS * D_ * 2);
  bf16_t* Ehat = (bf16_t*)alloc((size_t)MROWS * EHD * 2);
  bf16_t* WQt  = (bf16_t*)alloc((size_t)H_ * DH_ * D_ * 2);
  bf16_t* WKt  = (bf16_t*)alloc((size_t)H_ * DH_ * EHD * 2);
  bf16_t* WVt  = (bf16_t*)alloc((size_t)H_ * DH_ * EHD * 2);
  bf16_t* W0t  = (bf16_t*)alloc((size_t)D_ * HDH * 2);
  bf16_t* W1t  = (bf16_t*)alloc((size_t)FF_ * D_ * 2);
  bf16_t* W2t  = (bf16_t*)alloc((size_t)D_ * FF_ * 2);
  bf16_t* Qb   = (bf16_t*)alloc((size_t)H_ * MROWS * DH_ * 2);
  bf16_t* Kb   = (bf16_t*)alloc((size_t)H_ * MROWS * DH_ * 2);
  bf16_t* Vtb  = (bf16_t*)alloc((size_t)H_ * MROWS * DH_ * 2);
  bf16_t* Zc   = (bf16_t*)alloc((size_t)MROWS * HDH * 2);
  float*  zf   = (float*) alloc((size_t)MROWS * D_ * 4);
  bf16_t* zb   = (bf16_t*)alloc((size_t)MROWS * D_ * 2);
  bf16_t* f1   = (bf16_t*)alloc((size_t)MROWS * FF_ * 2);
  float*  hf   = (float*) alloc((size_t)MROWS * D_ * 4);

  auto cdiv = [](long a, long b) { return (int)((a + b - 1) / b); };

  // weight conversion/transposes
  k_transpose_bf16<<<dim3(cdiv((long)D_ * DH_, 256), 1, H_), 256, 0, stream>>>(WQ, WQt, D_, DH_);
  k_transpose_bf16<<<dim3(cdiv((long)EHD * DH_, 256), 1, H_), 256, 0, stream>>>(WK, WKt, EHD, DH_);
  k_transpose_bf16<<<dim3(cdiv((long)EHD * DH_, 256), 1, H_), 256, 0, stream>>>(WV, WVt, EHD, DH_);
  k_transpose_bf16<<<dim3(cdiv((long)HDH * D_, 256), 1, 1), 256, 0, stream>>>(W0, W0t, HDH, D_);
  k_transpose_bf16<<<dim3(cdiv((long)D_ * FF_, 256), 1, 1), 256, 0, stream>>>(ffn_W1, W1t, D_, FF_);
  k_transpose_bf16<<<dim3(cdiv((long)FF_ * D_, 256), 1, 1), 256, 0, stream>>>(ffn_W2, W2t, FF_, D_);

  // activations
  k_build_mhat<<<cdiv(MROWS * D_, 256), 256, 0, stream>>>(cate_x, targets, M_emb, pos_emb, Mhat);
  k_build_ehat<<<cdiv(MROWS * EHD, 256), 256, 0, stream>>>(cate_x, cont_x, E_emb, cate_emb, cont_W,
                                                           cont_b, Ehat);

  // Q = Mhat x WQ + bQ    (batched over heads, A shared)
  k_gemm<<<dim3(128, DH_ / 64, H_), 256, 0, stream>>>(
      Mhat, WQt, bQ, nullptr, nullptr, Qb, (int)MROWS, DH_, D_,
      0L, (long)DH_ * D_, (long)DH_, (long)MROWS * DH_, 0, 0);
  // K = Ehat x WK + bK
  k_gemm<<<dim3(128, DH_ / 64, H_), 256, 0, stream>>>(
      Ehat, WKt, bK, nullptr, nullptr, Kb, (int)MROWS, DH_, EHD,
      0L, (long)DH_ * EHD, (long)DH_, (long)MROWS * DH_, 0, 0);
  // V = Ehat x WV + bV   (stored transposed as [h][b][DH][SP])
  k_gemm<<<dim3(128, DH_ / 64, H_), 256, 0, stream>>>(
      Ehat, WVt, bV, nullptr, nullptr, Vtb, (int)MROWS, DH_, EHD,
      0L, (long)DH_ * EHD, (long)DH_, 0L, 0, 1);

  // fused attention -> Zc (B,SP,H*DH)
  k_attn<<<dim3(SP / 16, B_, H_), 128, 0, stream>>>(Qb, Kb, Vtb, Zc);

  // z = Zc x W0   (f32 + bf16 copies)
  k_gemm<<<dim3(128, D_ / 64, 1), 256, 0, stream>>>(
      Zc, W0t, nullptr, nullptr, zf, zb, (int)MROWS, D_, HDH, 0L, 0L, 0L, 0L, 0, 0);
  // f1 = relu(z x W1 + b1)
  k_gemm<<<dim3(128, FF_ / 64, 1), 256, 0, stream>>>(
      zb, W1t, ffn_b1, nullptr, nullptr, f1, (int)MROWS, FF_, D_, 0L, 0L, 0L, 0L, 1, 0);
  // h = z + (f1 x W2 + b2)
  k_gemm<<<dim3(128, D_ / 64, 1), 256, 0, stream>>>(
      f1, W2t, ffn_b2, zf, hf, nullptr, (int)MROWS, D_, FF_, 0L, 0L, 0L, 0L, 0, 0);

  // layernorm + final projection -> out (B,S)
  k_lnfinal<<<B_ * S_, 256, 0, stream>>>(hf, ln_g, ln_b, fin_W, fin_b, out);
}
